// KGIN_20701742367020
// MI455X (gfx1250) — compile-verified, matrix operand order checked by reading, not run on hardware
//
#include <hip/hip_runtime.h>

#define E_NEW      4096
#define E_STATIC   250000
#define E_TOTAL    (E_NEW + E_STATIC)        // 254096
#define NUM_ENT    100000
#define D_ENT      128
#define D_REL      64
#define N_REL      9
#define N_INTENTS  8
#define NUM_LAYERS 3
#define E_PAD      254240                    // E_TOTAL + 9*16, 16-aligned
#define NTILES     (E_PAD / 16)              // 15890
#define ENT_ELEMS  (NUM_ENT * D_ENT)         // 12,800,000

typedef __bf16 v16bf __attribute__((ext_vector_type(16)));
typedef float  v8f   __attribute__((ext_vector_type(8)));
union BF16x16 { v16bf v; unsigned short s[16]; uint4 q[2]; };

// ---------------- helpers ----------------

__device__ __forceinline__ unsigned short f2bf(float f) {
    unsigned int u = __float_as_uint(f);
    unsigned int r = u + 0x7FFFu + ((u >> 16) & 1u);   // round-to-nearest-even
    return (unsigned short)(r >> 16);
}

__device__ __forceinline__ float selu_f(float x) {
    const float a  = 1.6732632423543772f;
    const float sc = 1.0507009873554805f;
    return sc * (x > 0.f ? x : a * (__expf(x) - 1.f));
}

__device__ __forceinline__ int edge_field(const int* ne, const int* se, int e, int f) {
    return (e < E_NEW) ? ne[e * 5 + f] : se[(size_t)(e - E_NEW) * 5 + f];
}

// Fragment element mapping (16-bit A 16x32 / B 32x16, wave32), per ISA 7.12.2:
//   element t in lane L:  v = t/2
//   k_local = (v<4 ? 0 : 16) + (L>=16 ? 8 : 0) + (v&3)*2 + (t&1)
//   A row  = L%16 ;  B col = L%16
__device__ __forceinline__ int frag_k(int lane, int t) {
    int v = t >> 1;
    return ((v < 4) ? 0 : 16) + ((lane & 16) ? 8 : 0) + ((v & 3) << 1) + (t & 1);
}

// ---------------- setup kernels ----------------

__global__ void k_init_ent(float* ent, const float* def_emb, const int* new_types,
                           const float* ent_emb) {
    int i = blockIdx.x * blockDim.x + threadIdx.x;
    if (i >= ENT_ELEMS) return;
    int row = i / D_ENT, col = i - row * D_ENT;
    if (row < E_NEW) ent[i] = def_emb[new_types[row] * D_ENT + col];
    else             ent[i] = ent_emb[(size_t)(row - E_NEW) * D_ENT + col];
}

// bf16 mirror of the entity table (rebuilt each layer; feeds WMMA-A gathers + intent chain)
__global__ void k_ent2bf(const float* ent, unsigned short* entBF) {
    int i = blockIdx.x * blockDim.x + threadIdx.x;
    if (i < ENT_ELEMS) entBF[i] = f2bf(ent[i]);
}

// Pre-shuffle weights into WMMA-B fragment layout:
//   dst[(((r*NT + nt)*4 + kb)*32 + lane)*16 + t] = bf16(W[r][kb*32+frag_k(lane,t)][nt*16+lane%16])
__global__ void k_cvt_frag(const float* W, unsigned short* dst, int NT, int NC, int total) {
    int i = blockIdx.x * blockDim.x + threadIdx.x;
    if (i >= total) return;
    int t    = i & 15;
    int lane = (i >> 4) & 31;
    int kb   = (i >> 9) & 3;
    int rest = i >> 11;
    int nt   = rest % NT, r = rest / NT;
    int k = kb * 32 + frag_k(lane, t);
    int n = nt * 16 + (lane & 15);
    dst[i] = f2bf(W[((size_t)r * D_ENT + k) * NC + n]);
}

__global__ void k_fill_int(int* p, int n, int val) {
    int i = blockIdx.x * blockDim.x + threadIdx.x;
    if (i < n) p[i] = val;
}

__global__ void k_zero_f32(float* p, int n) {
    int i = blockIdx.x * blockDim.x + threadIdx.x;
    if (i < n) p[i] = 0.f;
}

// ctrl layout: [0..8]=hist, [9..17]=base, [18..26]=cursor
__global__ void k_hist(const int* ne, const int* se, int* ctrl) {
    int e = blockIdx.x * blockDim.x + threadIdx.x;
    if (e >= E_TOTAL) return;
    atomicAdd(&ctrl[edge_field(ne, se, e, 4)], 1);
}

__global__ void k_prefix(int* ctrl) {
    int total = 0;
    for (int r = 0; r < N_REL; ++r) {
        ctrl[9 + r] = total;
        total += ((ctrl[r] + 15) >> 4) << 4;   // pad each bin to 16
        ctrl[18 + r] = 0;
    }
}

__global__ void k_scatter(const int* ne, const int* se, int* ctrl, int* perm) {
    int e = blockIdx.x * blockDim.x + threadIdx.x;
    if (e >= E_TOTAL) return;
    int r = edge_field(ne, se, e, 4);
    int off = atomicAdd(&ctrl[18 + r], 1);
    perm[ctrl[9 + r] + off] = e;
}

// ---------------- per-edge intent chain (scalar VALU; ~4% of FLOPs) ----------------

__global__ void k_iw(const unsigned short* entBF, const int* ne, const int* se,
                     const float* Wint, const float* bint,
                     const float* Wibn, const float* bibn, float* iw) {
    int e = blockIdx.x * blockDim.x + threadIdx.x;
    if (e >= E_TOTAL) return;
    int tgt = edge_field(ne, se, e, 2);
    int tt  = edge_field(ne, se, e, 3);
    int rel = edge_field(ne, se, e, 4);
    const uint4* trow = (const uint4*)(entBF + (size_t)tgt * D_ENT);

    float acc[N_INTENTS];
    for (int j = 0; j < N_INTENTS; ++j) acc[j] = bint[tt * N_INTENTS + j];
    const float* W = Wint + (size_t)tt * D_ENT * N_INTENTS;
    for (int i8 = 0; i8 < 16; ++i8) {           // 8 bf16 per uint4
        uint4 p = trow[i8];
        float f[8];
        f[0] = __uint_as_float(p.x << 16); f[1] = __uint_as_float(p.x & 0xFFFF0000u);
        f[2] = __uint_as_float(p.y << 16); f[3] = __uint_as_float(p.y & 0xFFFF0000u);
        f[4] = __uint_as_float(p.z << 16); f[5] = __uint_as_float(p.z & 0xFFFF0000u);
        f[6] = __uint_as_float(p.w << 16); f[7] = __uint_as_float(p.w & 0xFFFF0000u);
        const float* Wi = W + (size_t)i8 * 8 * N_INTENTS;
        for (int u = 0; u < 8; ++u)
            for (int j = 0; j < N_INTENTS; ++j) acc[j] += f[u] * Wi[u * N_INTENTS + j];
    }
    float mx = acc[0];
    for (int j = 1; j < N_INTENTS; ++j) mx = fmaxf(mx, acc[j]);
    float sum = 0.f;
    for (int j = 0; j < N_INTENTS; ++j) { acc[j] = __expf(acc[j] - mx); sum += acc[j]; }
    float inv = 1.f / sum;
    for (int j = 0; j < N_INTENTS; ++j) acc[j] *= inv;

    float rw[N_REL];
    for (int k = 0; k < N_REL; ++k) rw[k] = bibn[tt * N_REL + k];
    const float* W2 = Wibn + (size_t)tt * N_INTENTS * N_REL;
    for (int j = 0; j < N_INTENTS; ++j)
        for (int k = 0; k < N_REL; ++k) rw[k] += acc[j] * W2[j * N_REL + k];
    mx = rw[0];
    for (int k = 1; k < N_REL; ++k) mx = fmaxf(mx, rw[k]);
    sum = 0.f;
    for (int k = 0; k < N_REL; ++k) { rw[k] = __expf(rw[k] - mx); sum += rw[k]; }
    iw[e] = rw[rel] / sum;
}

// ---------------- WMMA tile kernels ----------------

// Stage 16 gathered bf16 rows (256 B each) into LDS via async global->LDS DMA.
// Two rows per instruction: lanes 0-15 cover row 2i (16 B each), lanes 16-31 row 2i+1.
__device__ __forceinline__ void stage_A_async(unsigned short* A, const unsigned short* entBF,
                                              const int* sSrc, int lane) {
    for (int i = 0; i < 8; ++i) {
        int m = i * 2 + (lane >> 4);
        int sidx = sSrc[m];
        if (sidx < 0) sidx = 0;                 // padded rows: load row 0, masked later
        const unsigned short* gp = entBF + (size_t)sidx * D_ENT + (lane & 15) * 8;
        unsigned int lds = (unsigned int)(unsigned long long)(&A[m * D_ENT + (lane & 15) * 8]);
        asm volatile("global_load_async_to_lds_b128 %0, %1, off"
                     :: "v"(lds), "v"((unsigned long long)gp) : "memory");
    }
    asm volatile("s_wait_asynccnt 0x0" ::: "memory");
}

__device__ __forceinline__ void load_A_frags(BF16x16* aF, const unsigned short* A, int lane) {
    for (int kb = 0; kb < 4; ++kb)
        for (int t = 0; t < 16; ++t)
            aF[kb].s[t] = A[(lane & 15) * D_ENT + kb * 32 + frag_k(lane, t)];
}

// score kernel: src_rel = selu(src @ W_rel[r] + b_rel[r]); score = <src_rel, rel_e[r]> * iw
__global__ __launch_bounds__(32) void k_score(
        const unsigned short* entBF, const int* ne, const int* se, const int* perm,
        const unsigned short* WrelF, const float* b_rel, const float* rel_emb,
        const float* iw, float* exps, float* denom) {
    int tile = blockIdx.x, lane = threadIdx.x;
    __shared__ unsigned short A[16 * D_ENT];
    __shared__ float sScore[16];
    __shared__ int sE[16], sSrc[16], sRel;

    int e = perm[tile * 16 + (lane & 15)];
    if (lane < 16) {
        sE[lane] = e;
        sScore[lane] = 0.f;
        sSrc[lane] = (e >= 0) ? edge_field(ne, se, e, 0) : -1;
        if (e >= 0) sRel = edge_field(ne, se, e, 4);   // bin-uniform
    }
    __syncthreads();
    bool any = false;
    for (int m = 0; m < 16; ++m) any |= (sE[m] >= 0);
    if (!any) return;                                   // wave-uniform exit
    int r = sRel;

    stage_A_async(A, entBF, sSrc, lane);
    __syncthreads();

    BF16x16 aF[4];
    load_A_frags(aF, A, lane);

    for (int nt = 0; nt < 4; ++nt) {                    // 64 output cols
        v8f c = {};
        for (int kb = 0; kb < 4; ++kb) {                // K = 128
            const uint4* bp = (const uint4*)(WrelF +
                ((((size_t)r * 4 + nt) * 4 + kb) * 32 + lane) * 16);
            BF16x16 bF; bF.q[0] = bp[0]; bF.q[1] = bp[1];
            c = __builtin_amdgcn_wmma_f32_16x16x32_bf16(false, aF[kb].v, false, bF.v,
                                                        (short)0, c, false, false);
        }
        int N = nt * 16 + (lane & 15);
        float bn = b_rel[r * D_REL + N];
        float re = rel_emb[r * D_REL + N];
        for (int v = 0; v < 8; ++v) {
            int M = v + ((lane & 16) ? 8 : 0);
            float x = selu_f(c[v] + bn);
            atomicAdd(&sScore[M], x * re);              // ds_add_f32
        }
    }
    __syncthreads();
    if (lane < 16 && e >= 0) {
        float es = __expf(sScore[lane] * iw[e]);
        exps[e] = es;
        atomicAdd(&denom[edge_field(ne, se, e, 2)], es);
    }
}

__global__ void k_recip(const float* denom, float* recip) {
    int i = blockIdx.x * blockDim.x + threadIdx.x;
    if (i >= NUM_ENT) return;
    float d = denom[i];
    recip[i] = (d != 0.f) ? (1.f / d) : 0.f;
}

// message kernel: msg = dist * (src @ W_msg[r]); scatter-add into entOut; record dist
__global__ __launch_bounds__(32) void k_msg(
        const unsigned short* entBF, const int* ne, const int* se, const int* perm,
        const unsigned short* WmsgF, const float* exps, const float* recip,
        float* entOut, float* distL) {
    int tile = blockIdx.x, lane = threadIdx.x;
    __shared__ unsigned short A[16 * D_ENT];
    __shared__ float sDist[16];
    __shared__ int sE[16], sSrc[16], sTgt[16], sRel;

    int e = perm[tile * 16 + (lane & 15)];
    if (lane < 16) {
        sE[lane] = e;
        int t = -1; float dv = 0.f; int s = -1;
        if (e >= 0) {
            t = edge_field(ne, se, e, 2);
            s = edge_field(ne, se, e, 0);
            dv = exps[e] * recip[t];
            distL[e] = dv;
            sRel = edge_field(ne, se, e, 4);
        }
        sTgt[lane] = t; sSrc[lane] = s; sDist[lane] = dv;
    }
    __syncthreads();
    bool any = false;
    for (int m = 0; m < 16; ++m) any |= (sE[m] >= 0);
    if (!any) return;
    int r = sRel;

    stage_A_async(A, entBF, sSrc, lane);
    __syncthreads();

    BF16x16 aF[4];
    load_A_frags(aF, A, lane);

    for (int nt = 0; nt < 8; ++nt) {                    // 128 output cols
        v8f c = {};
        for (int kb = 0; kb < 4; ++kb) {
            const uint4* bp = (const uint4*)(WmsgF +
                ((((size_t)r * 8 + nt) * 4 + kb) * 32 + lane) * 16);
            BF16x16 bF; bF.q[0] = bp[0]; bF.q[1] = bp[1];
            c = __builtin_amdgcn_wmma_f32_16x16x32_bf16(false, aF[kb].v, false, bF.v,
                                                        (short)0, c, false, false);
        }
        int col = nt * 16 + (lane & 15);
        for (int v = 0; v < 8; ++v) {
            int M = v + ((lane & 16) ? 8 : 0);
            int t = sTgt[M];
            if (t >= 0) atomicAdd(&entOut[(size_t)t * D_ENT + col], c[v] * sDist[M]);
        }
    }
}

// ---------------- host launch ----------------

extern "C" void kernel_launch(void* const* d_in, const int* in_sizes, int n_in,
                              void* d_out, int out_size, void* d_ws, size_t ws_size,
                              hipStream_t stream) {
    const int*   ne        = (const int*)d_in[0];
    const int*   new_types = (const int*)d_in[1];
    const int*   se        = (const int*)d_in[2];
    const float* ent_emb   = (const float*)d_in[3];
    const float* def_emb   = (const float*)d_in[4];
    const float* rel_emb   = (const float*)d_in[5];
    const float* W_rel     = (const float*)d_in[6];
    const float* b_rel     = (const float*)d_in[7];
    const float* W_msg     = (const float*)d_in[8];
    const float* W_int     = (const float*)d_in[9];
    const float* b_int     = (const float*)d_in[10];
    const float* W_ibn     = (const float*)d_in[11];
    const float* b_ibn     = (const float*)d_in[12];

    char* ws = (char*)d_ws;
    size_t off = 0;
    float*          entA   = (float*)(ws + off);          off += (size_t)ENT_ELEMS * 4;
    unsigned short* entBF  = (unsigned short*)(ws + off); off += (size_t)ENT_ELEMS * 2;
    unsigned short* WrelF  = (unsigned short*)(ws + off); off += (size_t)N_REL * D_ENT * D_REL * 2;
    unsigned short* WmsgF  = (unsigned short*)(ws + off); off += (size_t)N_REL * D_ENT * D_ENT * 2;
    int*            perm   = (int*)(ws + off);            off += (size_t)E_PAD * 4;
    float*          iw     = (float*)(ws + off);          off += (size_t)E_TOTAL * 4;
    float*          exps   = (float*)(ws + off);          off += (size_t)E_TOTAL * 4;
    float*          denom  = (float*)(ws + off);          off += (size_t)NUM_ENT * 4;
    float*          recip  = (float*)(ws + off);          off += (size_t)NUM_ENT * 4;
    int*            ctrl   = (int*)(ws + off);            off += 27 * 4;

    float* out     = (float*)d_out;
    float* entB    = out;                      // d_out ent region doubles as ping-pong buffer
    float* distOut = out + (size_t)ENT_ELEMS;  // 3 x E_TOTAL layer edge weights

    const int TB = 256;
    const int gE   = (E_TOTAL + TB - 1) / TB;
    const int gEnt = (ENT_ELEMS + TB - 1) / TB;
    const int gNE  = (NUM_ENT + TB - 1) / TB;

    // one-time setup
    k_init_ent<<<gEnt, TB, 0, stream>>>(entA, def_emb, new_types, ent_emb);
    k_cvt_frag<<<(N_REL*D_ENT*D_REL + TB - 1)/TB, TB, 0, stream>>>(
        W_rel, WrelF, 4, D_REL, N_REL*D_ENT*D_REL);
    k_cvt_frag<<<(N_REL*D_ENT*D_ENT + TB - 1)/TB, TB, 0, stream>>>(
        W_msg, WmsgF, 8, D_ENT, N_REL*D_ENT*D_ENT);
    k_fill_int<<<(E_PAD + TB - 1)/TB, TB, 0, stream>>>(perm, E_PAD, -1);
    k_fill_int<<<1, 32, 0, stream>>>(ctrl, 27, 0);
    k_hist<<<gE, TB, 0, stream>>>(ne, se, ctrl);
    k_prefix<<<1, 1, 0, stream>>>(ctrl);
    k_scatter<<<gE, TB, 0, stream>>>(ne, se, ctrl, perm);

    for (int l = 0; l < NUM_LAYERS; ++l) {
        float* entIn  = (l & 1) ? entB : entA;
        float* entOut = (l & 1) ? entA : entB;
        k_ent2bf<<<gEnt, TB, 0, stream>>>(entIn, entBF);        // bf16 mirror for gathers
        k_zero_f32<<<gEnt, TB, 0, stream>>>(entOut, ENT_ELEMS);
        k_zero_f32<<<gNE, TB, 0, stream>>>(denom, NUM_ENT);
        k_iw<<<gE, TB, 0, stream>>>(entBF, ne, se, W_int, b_int, W_ibn, b_ibn, iw);
        k_score<<<NTILES, 32, 0, stream>>>(entBF, ne, se, perm, WrelF, b_rel, rel_emb,
                                           iw, exps, denom);
        k_recip<<<gNE, TB, 0, stream>>>(denom, recip);
        k_msg<<<NTILES, 32, 0, stream>>>(entBF, ne, se, perm, WmsgF, exps, recip,
                                         entOut, distOut + (size_t)l * E_TOTAL);
    }
    // layer 2 writes entB == d_out ent region; dist written per-layer above.
}
